// GaussianSpikingEncoder_76819785056378
// MI455X (gfx1250) — matrix operand order repaired and checked by reading, not compiled, not run on hardware
//
#include <hip/hip_runtime.h>
#include <hip/hip_bf16.h>

typedef __attribute__((ext_vector_type(16))) _Float16 v16h;
typedef __attribute__((ext_vector_type(8)))  _Float16 h8_t;
typedef __attribute__((ext_vector_type(8)))  float    v8f;

#define TSTEPS 4
#define BATCH  8
#define NPTS   2048
#define BN     16384      // BATCH*NPTS
#define KNN    16
#define BNK    262144     // BN*KNN
#define KDMAX  256        // max inner dim staged in LDS

// ---------------------------------------------------------------------------
// WMMA fragment helpers (CDNA5 16x16x32 f16 -> f32)
// A-matrix 16x32: lane m = lane&15; hi = lane>>4. 16 contiguous K starting at hi*16
// (matches B)   => two contiguous 8-half (16B) runs.
// B-matrix 32x16: lane n = lane&15; 16 contiguous K starting at hi*16 (one 32B load).
// C/D 16x16 f32: element v -> (m = hi*8+v, n = lane&15).
// ---------------------------------------------------------------------------
__device__ __forceinline__ v16h wmma_ld_a(const _Float16* base, int hi) {
  h8_t lo = *reinterpret_cast<const h8_t*>(base + hi * 16);
  h8_t hh = *reinterpret_cast<const h8_t*>(base + hi * 16 + 8);
  v16h r;
#pragma unroll
  for (int i = 0; i < 8; ++i) { r[i] = lo[i]; r[8 + i] = hh[i]; }
  return r;
}

__device__ __forceinline__ v16h wmma_ld_b(const _Float16* p) {
  return *reinterpret_cast<const v16h*>(p);
}

__device__ __forceinline__ v8f wmma_f16f32(v16h a, v16h b, v8f c) {
  return __builtin_amdgcn_wmma_f32_16x16x32_f16(false, a, false, b, (short)0, c,
                                                false, false);
}

// Async Global -> LDS copy, 16 bytes per enabled lane (tracked by ASYNCcnt).
__device__ __forceinline__ void async_g2l_b128(unsigned lds_off,
                                               const void* gaddr) {
  asm volatile("global_load_async_to_lds_b128 %0, %1, off"
               :
               : "v"(lds_off), "v"(gaddr)
               : "memory");
}
__device__ __forceinline__ void wait_async0() {
  asm volatile("s_wait_asynccnt 0" ::: "memory");
}

__device__ __forceinline__ float lanesum16(float v) {
#pragma unroll
  for (int m = 1; m < 16; m <<= 1) v += __shfl_xor(v, m, 32);
  return v;
}
__device__ __forceinline__ float lanemax16(float v) {
#pragma unroll
  for (int m = 1; m < 16; m <<= 1) v = fmaxf(v, __shfl_xor(v, m, 32));
  return v;
}

enum {
  MODE_RAW_STATS      = 0,  // TT=1: store raw pre-act (half) + accumulate per-d stats
  MODE_LIF_STORE      = 1,  // +bias, LIF over t, store spikes per t
  MODE_LIF_MAXAT      = 2,  // +bias, LIF, max over 16 cols, atomicMax into (T,B,M) f32
  MODE_STATS          = 3,  // accumulate per-(t,d) stats only
  MODE_BN_LIF_STORE   = 4,  // BN(t,d), LIF, store spikes per t
  MODE_BN_LIF_MAXSTORE= 5,  // BN, LIF, max over 16 cols (=k group), store pooled spikes
  MODE_BN_LIF_MAXAT   = 6   // BN, LIF, max over 16 cols, atomicMax into (T,B,M) f32
};

// ---------------------------------------------------------------------------
// Mode-specific epilogue shared by both GEMM kernels (per-wave, uniform flow).
// ---------------------------------------------------------------------------
template <int TT, int MODE>
__device__ __forceinline__ void gemm_tail(
    v8f (&acc)[TT], int lo16, long long ct, int mbase, int M,
    const float* __restrict__ bias, const float* __restrict__ bn_mean,
    const float* __restrict__ bn_inv, const float* __restrict__ bn_g,
    const float* __restrict__ bn_b, float* __restrict__ st_sum,
    float* __restrict__ st_sq, _Float16* __restrict__ outH,
    long long out_tstride, float* __restrict__ outF, int colsPerGroup) {
  if (MODE == MODE_STATS || MODE == MODE_RAW_STATS) {
#pragma unroll
    for (int t = 0; t < TT; ++t)
#pragma unroll
      for (int v = 0; v < 8; ++v) {
        float a  = acc[t][v];
        float s1 = lanesum16(a);
        float s2 = lanesum16(a * a);
        if (lo16 == 0) {
          atomicAdd(&st_sum[t * M + mbase + v], s1);
          atomicAdd(&st_sq [t * M + mbase + v], s2);
        }
      }
    if (MODE == MODE_RAW_STATS) {
      h8_t hv;
#pragma unroll
      for (int v = 0; v < 8; ++v) hv[v] = (_Float16)acc[0][v];
      *reinterpret_cast<h8_t*>(outH + (ct + lo16) * (long long)M + mbase) = hv;
    }
    return;
  }

  if (MODE == MODE_LIF_STORE || MODE == MODE_LIF_MAXAT) {
#pragma unroll
    for (int v = 0; v < 8; ++v) {
      float bv = bias[mbase + v];
#pragma unroll
      for (int t = 0; t < TT; ++t) acc[t][v] += bv;
    }
  }
  if (MODE == MODE_BN_LIF_STORE || MODE == MODE_BN_LIF_MAXSTORE ||
      MODE == MODE_BN_LIF_MAXAT) {
#pragma unroll
    for (int t = 0; t < TT; ++t)
#pragma unroll
      for (int v = 0; v < 8; ++v) {
        int im = t * M + mbase + v;
        acc[t][v] = (acc[t][v] - bn_mean[im]) * bn_inv[im] * bn_g[im] + bn_b[im];
      }
  }

  // LIF over t in registers: v' = (v + x)/2 ; spike at v'>=1 ; reset on spike.
  float sp[TT][8];
#pragma unroll
  for (int v = 0; v < 8; ++v) {
    float vm = 0.0f;
#pragma unroll
    for (int t = 0; t < TT; ++t) {
      vm = 0.5f * (vm + acc[t][v]);
      float s = (vm >= 1.0f) ? 1.0f : 0.0f;
      sp[t][v] = s;
      vm *= (1.0f - s);
    }
  }

  if (MODE == MODE_LIF_STORE || MODE == MODE_BN_LIF_STORE) {
#pragma unroll
    for (int t = 0; t < TT; ++t) {
      h8_t hv;
#pragma unroll
      for (int v = 0; v < 8; ++v) hv[v] = (_Float16)sp[t][v];
      *reinterpret_cast<h8_t*>(outH + (long long)t * out_tstride +
                               (ct + lo16) * (long long)M + mbase) = hv;
    }
  }
  if (MODE == MODE_LIF_MAXAT || MODE == MODE_BN_LIF_MAXAT) {
    int g = (int)(ct / colsPerGroup);
#pragma unroll
    for (int t = 0; t < TT; ++t)
#pragma unroll
      for (int v = 0; v < 8; ++v) {
        float mx = lanemax16(sp[t][v]);
        if (lo16 == 0)
          atomicMax((int*)&outF[((long long)(t * BATCH + g)) * M + mbase + v],
                    __float_as_int(mx));
      }
  }
  if (MODE == MODE_BN_LIF_MAXSTORE) {
    long long bn = ct >> 4;  // 16-column tile == one (b,n) neighbor group
#pragma unroll
    for (int t = 0; t < TT; ++t) {
      h8_t hv;
#pragma unroll
      for (int v = 0; v < 8; ++v) hv[v] = (_Float16)lanemax16(sp[t][v]);
      if (lo16 == 0)
        *reinterpret_cast<h8_t*>(outH + (long long)t * out_tstride +
                                 bn * (long long)M + mbase) = hv;
    }
  }
}

// ---------------------------------------------------------------------------
// Single-wave GEMM (used for the tiny 28->64 edge conv).
// ---------------------------------------------------------------------------
template <int TT, int MODE>
__global__ void __launch_bounds__(32)
gemm_wmma(const _Float16* __restrict__ W, const _Float16* __restrict__ Bact,
          long long b_tstride, int KDIM, int M,
          const float* __restrict__ bias,
          const float* __restrict__ bn_mean, const float* __restrict__ bn_inv,
          const float* __restrict__ bn_g, const float* __restrict__ bn_b,
          float* __restrict__ st_sum, float* __restrict__ st_sq,
          _Float16* __restrict__ outH, long long out_tstride,
          float* __restrict__ outF, int colsPerGroup) {
  const int lane = threadIdx.x;
  const int hi = lane >> 4, lo16 = lane & 15;
  const long long ct = (long long)blockIdx.x * 16;
  const int mt = blockIdx.y * 16;

  v8f acc[TT];
#pragma unroll
  for (int t = 0; t < TT; ++t)
#pragma unroll
    for (int v = 0; v < 8; ++v) acc[t][v] = 0.0f;

  const _Float16* Arow = W + (long long)(mt + lo16) * KDIM;
  const _Float16* Bcol = Bact + (ct + lo16) * (long long)KDIM;

  for (int k0 = 0; k0 < KDIM; k0 += 32) {
    v16h a = wmma_ld_a(Arow + k0, hi);
#pragma unroll
    for (int t = 0; t < TT; ++t) {
      v16h b = wmma_ld_b(Bcol + (long long)t * b_tstride + k0 + hi * 16);
      acc[t] = wmma_f16f32(a, b, acc[t]);
    }
  }
  gemm_tail<TT, MODE>(acc, lo16, ct, mt + hi * 8, M, bias, bn_mean, bn_inv,
                      bn_g, bn_b, st_sum, st_sq, outH, out_tstride, outF,
                      colsPerGroup);
}

// ---------------------------------------------------------------------------
// Workgroup GEMM: 8 waves = 128-row M-chunk sharing one 16-column activation
// tile (all T planes) async-staged into LDS (global_load_async_to_lds_b128,
// ASYNCcnt). Each wave then runs the K-loop out of LDS with its own A rows.
// ---------------------------------------------------------------------------
template <int TT, int MODE>
__global__ void __launch_bounds__(256)
gemm_wmma_wg(const _Float16* __restrict__ W, const _Float16* __restrict__ Bact,
             long long b_tstride, int KDIM, int M,
             const float* __restrict__ bias,
             const float* __restrict__ bn_mean, const float* __restrict__ bn_inv,
             const float* __restrict__ bn_g, const float* __restrict__ bn_b,
             float* __restrict__ st_sum, float* __restrict__ st_sq,
             _Float16* __restrict__ outH, long long out_tstride,
             float* __restrict__ outF, int colsPerGroup) {
  __shared__ __attribute__((aligned(64))) _Float16 sB[TSTEPS * 16 * KDMAX];

  const int tid = threadIdx.x;
  const long long ct = (long long)blockIdx.x * 16;

  // Stage B tile: TT planes x 16 cols x KDIM halves, contiguous per plane.
  {
    const unsigned sb0 = (unsigned)(uintptr_t)&sB[0];
    const int chunksPerT = 2 * KDIM;  // 16-byte chunks per plane
    const int total = TT * chunksPerT;
    for (int c = tid; c < total; c += 256) {
      int t = c / chunksPerT;
      int j = c - t * chunksPerT;
      const _Float16* src =
          Bact + (long long)t * b_tstride + ct * (long long)KDIM + j * 8;
      unsigned dst = sb0 + (unsigned)((t * 16 * KDIM + j * 8) * 2);
      async_g2l_b128(dst, (const void*)src);
    }
    wait_async0();
    __syncthreads();
  }

  const int wv = tid >> 5, lane = tid & 31;
  const int hi = lane >> 4, lo16 = lane & 15;
  const int mt = blockIdx.y * 128 + wv * 16;

  v8f acc[TT];
#pragma unroll
  for (int t = 0; t < TT; ++t)
#pragma unroll
    for (int v = 0; v < 8; ++v) acc[t][v] = 0.0f;

  const _Float16* Arow = W + (long long)(mt + lo16) * KDIM;

  for (int k0 = 0; k0 < KDIM; k0 += 32) {
    if (k0 + 32 < KDIM) __builtin_prefetch(Arow + k0 + 32, 0, 1);
    v16h a = wmma_ld_a(Arow + k0, hi);
#pragma unroll
    for (int t = 0; t < TT; ++t) {
      v16h b =
          *reinterpret_cast<const v16h*>(&sB[(t * 16 + lo16) * KDIM + k0 + hi * 16]);
      acc[t] = wmma_f16f32(a, b, acc[t]);
    }
  }
  gemm_tail<TT, MODE>(acc, lo16, ct, mt + hi * 8, M, bias, bn_mean, bn_inv,
                      bn_g, bn_b, st_sum, st_sq, outH, out_tstride, outF,
                      colsPerGroup);
}

// ---------------------------------------------------------------------------
__global__ void k_zero_f32(float* p, long long n) {
  long long i = (long long)blockIdx.x * blockDim.x + threadIdx.x;
  long long stride = (long long)gridDim.x * blockDim.x;
  for (; i < n; i += stride) p[i] = 0.0f;
}

__global__ void k_cvt_w(const float* __restrict__ src, _Float16* __restrict__ dst,
                        int M, int Ks, int Kp) {
  int i = blockIdx.x * 256 + threadIdx.x;
  if (i >= M * Kp) return;
  int m = i / Kp, k = i - m * Kp;
  dst[i] = (k < Ks) ? (_Float16)src[m * Ks + k] : (_Float16)0.0f;
}

// T-Net conv1 (3->64) + LIF (input identical across t) -> spikes s1 (half)
__global__ void k_tnet1(const float* __restrict__ x, const float* __restrict__ w,
                        const float* __restrict__ b, _Float16* __restrict__ s1,
                        long long tstride) {
  int i = blockIdx.x * 256 + threadIdx.x;  // BN*64
  if (i >= BN * 64) return;
  int col = i >> 6, d = i & 63;
  int bb = col >> 11, n = col & (NPTS - 1);
  const float* xb = x + (long long)bb * 14 * NPTS;
  float acc = b[d];
#pragma unroll
  for (int c = 0; c < 3; ++c) acc += xb[c * NPTS + n] * w[d * 3 + c];
  float vm = 0.0f;
#pragma unroll
  for (int t = 0; t < TSTEPS; ++t) {
    vm = 0.5f * (vm + acc);
    float s = (vm >= 1.0f) ? 1.0f : 0.0f;
    s1[(long long)t * tstride + (long long)col * 64 + d] = (_Float16)s;
    vm *= (1.0f - s);
  }
}

// T-Net FC chain per batch: 1024->512->256->9, LIF between, mean over t, +I.
__global__ void __launch_bounds__(512)
k_tnet_fc(const float* __restrict__ h, const float* __restrict__ fw1,
          const float* __restrict__ fb1, const float* __restrict__ fw2,
          const float* __restrict__ fb2, const float* __restrict__ fw3,
          const float* __restrict__ fb3, float* __restrict__ trans) {
  __shared__ float s1[TSTEPS][512];
  __shared__ float s2[TSTEPS][256];
  int b = blockIdx.x, tid = threadIdx.x;
  {
    float acc[TSTEPS];
#pragma unroll
    for (int t = 0; t < TSTEPS; ++t) acc[t] = fb1[tid];
    for (int i = 0; i < 1024; ++i) {
      float w = fw1[tid * 1024 + i];
#pragma unroll
      for (int t = 0; t < TSTEPS; ++t)
        acc[t] += h[((long long)(t * BATCH + b)) * 1024 + i] * w;
    }
    float vm = 0.0f;
#pragma unroll
    for (int t = 0; t < TSTEPS; ++t) {
      vm = 0.5f * (vm + acc[t]);
      float s = (vm >= 1.0f) ? 1.0f : 0.0f;
      s1[t][tid] = s;
      vm *= (1.0f - s);
    }
  }
  __syncthreads();
  if (tid < 256) {
    float acc[TSTEPS];
#pragma unroll
    for (int t = 0; t < TSTEPS; ++t) acc[t] = fb2[tid];
    for (int i = 0; i < 512; ++i) {
      float w = fw2[tid * 512 + i];
#pragma unroll
      for (int t = 0; t < TSTEPS; ++t) acc[t] += s1[t][i] * w;
    }
    float vm = 0.0f;
#pragma unroll
    for (int t = 0; t < TSTEPS; ++t) {
      vm = 0.5f * (vm + acc[t]);
      float s = (vm >= 1.0f) ? 1.0f : 0.0f;
      s2[t][tid] = s;
      vm *= (1.0f - s);
    }
  }
  __syncthreads();
  if (tid < 9) {
    float acc = 0.0f;
#pragma unroll
    for (int t = 0; t < TSTEPS; ++t) {
      float a = fb3[tid];
      for (int i = 0; i < 256; ++i) a += s2[t][i] * fw3[tid * 256 + i];
      acc += a;
    }
    acc *= 0.25f;
    int r = tid / 3, c = tid - r * 3;
    trans[b * 9 + tid] = acc + (r == c ? 1.0f : 0.0f);
  }
}

// pos_a[b,d,n] = sum_c pos[b,c,n] * trans[b, c*3+d]
__global__ void k_transform(const float* __restrict__ x,
                            const float* __restrict__ trans,
                            float* __restrict__ pos_a) {
  int i = blockIdx.x * 256 + threadIdx.x;  // BN
  if (i >= BN) return;
  int b = i >> 11, n = i & (NPTS - 1);
  const float* xb = x + (long long)b * 14 * NPTS;
  float p0 = xb[n], p1 = xb[NPTS + n], p2 = xb[2 * NPTS + n];
  const float* T = trans + b * 9;
#pragma unroll
  for (int d = 0; d < 3; ++d)
    pos_a[((long long)(b * 3 + d)) * NPTS + n] =
        p0 * T[d] + p1 * T[3 + d] + p2 * T[6 + d];
}

// kNN: per (b, query) insertion top-16 of neg squared distance.
__global__ void __launch_bounds__(256)
k_knn(const float* __restrict__ pos_a, int* __restrict__ idx) {
  __shared__ float px[NPTS], py[NPTS], pz[NPTS];
  int b = blockIdx.x, tid = threadIdx.x;
  for (int i = tid; i < NPTS; i += 256) {
    px[i] = pos_a[((long long)(b * 3 + 0)) * NPTS + i];
    py[i] = pos_a[((long long)(b * 3 + 1)) * NPTS + i];
    pz[i] = pos_a[((long long)(b * 3 + 2)) * NPTS + i];
  }
  __syncthreads();
  int n = blockIdx.y * 256 + tid;
  float best[KNN];
  int bidx[KNN];
#pragma unroll
  for (int j = 0; j < KNN; ++j) { best[j] = -3.4e38f; bidx[j] = 0; }
  float qx = px[n], qy = py[n], qz = pz[n];
  for (int m = 0; m < NPTS; ++m) {
    float dx = px[m] - qx, dy = py[m] - qy, dz = pz[m] - qz;
    float nd = -(dx * dx + dy * dy + dz * dz);
    if (nd > best[KNN - 1]) {
      int j = KNN - 1;
      while (j > 0 && nd > best[j - 1]) {
        best[j] = best[j - 1];
        bidx[j] = bidx[j - 1];
        --j;
      }
      best[j] = nd;
      bidx[j] = m;
    }
  }
  long long base = ((long long)b * NPTS + n) * KNN;
  for (int j = 0; j < KNN; ++j) idx[base + j] = bidx[j];
}

// Build edge-feature activation matrix: 28 channels (ctr, nbr-ctr), padded to 32.
__global__ void k_gather(const float* __restrict__ x,
                         const float* __restrict__ pos_a,
                         const int* __restrict__ idx, _Float16* __restrict__ Act) {
  long long i = (long long)blockIdx.x * 256 + threadIdx.x;  // BNK
  if (i >= BNK) return;
  long long bn = i >> 4;
  int b = (int)(bn >> 11), n = (int)(bn & (NPTS - 1));
  int m = idx[i];
  _Float16* o = Act + i * 32;
  float ctr[14], nb[14];
#pragma unroll
  for (int c = 0; c < 3; ++c) {
    const float* p = pos_a + ((long long)(b * 3 + c)) * NPTS;
    ctr[c] = p[n];
    nb[c] = p[m];
  }
#pragma unroll
  for (int c = 3; c < 14; ++c) {
    const float* p = x + ((long long)b * 14 + c) * NPTS;
    ctr[c] = p[n];
    nb[c] = p[m];
  }
#pragma unroll
  for (int c = 0; c < 14; ++c) o[c] = (_Float16)ctr[c];
#pragma unroll
  for (int c = 0; c < 14; ++c) o[14 + c] = (_Float16)(nb[c] - ctr[c]);
  o[28] = o[29] = o[30] = o[31] = (_Float16)0.0f;
}

// Edge conv1 epilogue: xhat*(g,be) per t, LIF -> spikes s_e1.
__global__ void k_lif_e1(const _Float16* __restrict__ y,
                         const float* __restrict__ mean_,
                         const float* __restrict__ inv_,
                         const float* __restrict__ g, const float* __restrict__ be,
                         _Float16* __restrict__ s_e1, long long tstride) {
  long long i = (long long)blockIdx.x * 256 + threadIdx.x;  // BNK
  if (i >= BNK) return;
  const _Float16* yc = y + i * 64;
  for (int d = 0; d < 64; ++d) {
    float xh = ((float)yc[d] - mean_[d]) * inv_[d];
    float vm = 0.0f;
#pragma unroll
    for (int t = 0; t < TSTEPS; ++t) {
      float z = xh * g[t * 64 + d] + be[t * 64 + d];
      vm = 0.5f * (vm + z);
      float s = (vm >= 1.0f) ? 1.0f : 0.0f;
      s_e1[(long long)t * tstride + i * 64 + d] = (_Float16)s;
      vm *= (1.0f - s);
    }
  }
}

__global__ void k_stats_fin(const float* __restrict__ ssum,
                            const float* __restrict__ ssq, float* __restrict__ mean_,
                            float* __restrict__ inv_, int n, float cnt) {
  int i = blockIdx.x * 256 + threadIdx.x;
  if (i >= n) return;
  float mu = ssum[i] / cnt;
  float var = ssq[i] / cnt - mu * mu;
  mean_[i] = mu;
  inv_[i] = rsqrtf(var + 1e-5f);
}

__global__ void k_final(const float* __restrict__ hfinal, float* __restrict__ out) {
  int i = blockIdx.x * 256 + threadIdx.x;  // 8192
  if (i >= BATCH * 1024) return;
  float a = 0.0f;
#pragma unroll
  for (int t = 0; t < TSTEPS; ++t) a += hfinal[(long long)t * BATCH * 1024 + i];
  out[i] = 0.25f * a;
}

// ---------------------------------------------------------------------------
extern "C" void kernel_launch(void* const* d_in, const int* in_sizes, int n_in,
                              void* d_out, int out_size, void* d_ws, size_t ws_size,
                              hipStream_t stream) {
  const float* x     = (const float*)d_in[0];
  const float* t_w1  = (const float*)d_in[1];
  const float* t_b1  = (const float*)d_in[2];
  const float* t_w2  = (const float*)d_in[3];
  const float* t_b2  = (const float*)d_in[4];
  const float* t_w3  = (const float*)d_in[5];
  const float* t_b3  = (const float*)d_in[6];
  const float* t_fw1 = (const float*)d_in[7];
  const float* t_fb1 = (const float*)d_in[8];
  const float* t_fw2 = (const float*)d_in[9];
  const float* t_fb2 = (const float*)d_in[10];
  const float* t_fw3 = (const float*)d_in[11];
  const float* t_fb3 = (const float*)d_in[12];
  const float* w1    = (const float*)d_in[13];
  const float* w2    = (const float*)d_in[14];
  const float* w3    = (const float*)d_in[15];
  const float* w4    = (const float*)d_in[16];
  const float* g1    = (const float*)d_in[17];
  const float* be1   = (const float*)d_in[18];
  const float* g2    = (const float*)d_in[19];
  const float* be2   = (const float*)d_in[20];
  const float* g3    = (const float*)d_in[21];
  const float* be3   = (const float*)d_in[22];
  const float* g4    = (const float*)d_in[23];
  const float* be4   = (const float*)d_in[24];
  float* out = (float*)d_out;

  char* ws = (char*)d_ws;
  size_t off = 0;
  auto alloc = [&](size_t bytes) -> char* {
    char* p = ws + off;
    off += (bytes + 255) & ~(size_t)255;
    return p;
  };

  // zero-init region (atomic targets + stats), allocated contiguously first
  size_t zoff0 = off;
  float* h3max  = (float*)alloc(TSTEPS * BATCH * 1024 * 4);
  float* hfinal = (float*)alloc(TSTEPS * BATCH * 1024 * 4);
  float* stE1 = (float*)alloc(4 * 64 * 4);        // sum, sq, mean, inv
  float* stE2 = (float*)alloc(4 * 512 * 4);       // per (t,d) d<128
  float* stW3 = (float*)alloc(4 * 1024 * 4);      // per (t,d) d<256
  float* stW4 = (float*)alloc(4 * 4096 * 4);      // per (t,d) d<1024
  size_t zoff1 = off;

  float* pos_a = (float*)alloc(BATCH * 3 * NPTS * 4);
  float* trans = (float*)alloc(BATCH * 9 * 4);
  int* knn_idx = (int*)alloc((size_t)BNK * 4);

  _Float16* w2h  = (_Float16*)alloc(128 * 64 * 2);
  _Float16* w3h  = (_Float16*)alloc(1024 * 128 * 2);
  _Float16* e1h  = (_Float16*)alloc(64 * 32 * 2);
  _Float16* e2h  = (_Float16*)alloc(128 * 64 * 2);
  _Float16* w3eh = (_Float16*)alloc(256 * 128 * 2);
  _Float16* w4h  = (_Float16*)alloc(1024 * 256 * 2);

  _Float16* s1   = (_Float16*)alloc((size_t)TSTEPS * BN * 64 * 2);
  _Float16* s2   = (_Float16*)alloc((size_t)TSTEPS * BN * 128 * 2);
  _Float16* Act  = (_Float16*)alloc((size_t)BNK * 32 * 2);
  _Float16* y_e1 = (_Float16*)alloc((size_t)BNK * 64 * 2);
  _Float16* s_e1 = (_Float16*)alloc((size_t)TSTEPS * BNK * 64 * 2);
  _Float16* h2   = (_Float16*)alloc((size_t)TSTEPS * BN * 128 * 2);
  _Float16* s3   = (_Float16*)alloc((size_t)TSTEPS * BN * 256 * 2);

  // 1) zero atomic/stat region
  long long nz = (long long)(zoff1 - zoff0) / 4;
  k_zero_f32<<<64, 256, 0, stream>>>((float*)(ws + zoff0), nz);

  // 2) weight conversions (f32 row-major -> half row-major, K padded)
  k_cvt_w<<<(128 * 64 + 255) / 256, 256, 0, stream>>>(t_w2, w2h, 128, 64, 64);
  k_cvt_w<<<(1024 * 128 + 255) / 256, 256, 0, stream>>>(t_w3, w3h, 1024, 128, 128);
  k_cvt_w<<<(64 * 32 + 255) / 256, 256, 0, stream>>>(w1, e1h, 64, 28, 32);
  k_cvt_w<<<(128 * 64 + 255) / 256, 256, 0, stream>>>(w2, e2h, 128, 64, 64);
  k_cvt_w<<<(256 * 128 + 255) / 256, 256, 0, stream>>>(w3, w3eh, 256, 128, 128);
  k_cvt_w<<<(1024 * 256 + 255) / 256, 256, 0, stream>>>(w4, w4h, 1024, 256, 256);

  // 3) T-Net conv1 + LIF -> s1
  k_tnet1<<<(BN * 64 + 255) / 256, 256, 0, stream>>>(x, t_w1, t_b1, s1,
                                                     (long long)BN * 64);
  // 4) T-Net conv2 (64->128) + LIF -> s2
  gemm_wmma_wg<TSTEPS, MODE_LIF_STORE><<<dim3(BN / 16, 1), 256, 0, stream>>>(
      w2h, s1, (long long)BN * 64, 64, 128, t_b2, nullptr, nullptr, nullptr,
      nullptr, nullptr, nullptr, s2, (long long)BN * 128, nullptr, 0);
  // 5) T-Net conv3 (128->1024) + LIF + max over n -> h3max
  gemm_wmma_wg<TSTEPS, MODE_LIF_MAXAT><<<dim3(BN / 16, 8), 256, 0, stream>>>(
      w3h, s2, (long long)BN * 128, 128, 1024, t_b3, nullptr, nullptr, nullptr,
      nullptr, nullptr, nullptr, nullptr, 0, h3max, NPTS);
  // 6) T-Net FC chain -> trans (B,3,3)+I
  k_tnet_fc<<<BATCH, 512, 0, stream>>>(h3max, t_fw1, t_fb1, t_fw2, t_fb2, t_fw3,
                                       t_fb3, trans);
  // 7) apply transform
  k_transform<<<(BN + 255) / 256, 256, 0, stream>>>(x, trans, pos_a);
  // 8) kNN top-16
  k_knn<<<dim3(BATCH, NPTS / 256), 256, 0, stream>>>(pos_a, knn_idx);
  // 9) edge features -> Act (BNK x 32 half)
  k_gather<<<(BNK + 255) / 256, 256, 0, stream>>>(x, pos_a, knn_idx, Act);
  // 10) edge conv1 (28->64): raw pre-act + stats (t-independent), single-wave
  gemm_wmma<1, MODE_RAW_STATS><<<dim3(BNK / 16, 4), 32, 0, stream>>>(
      e1h, Act, 0, 32, 64, nullptr, nullptr, nullptr, nullptr, nullptr,
      stE1 + 0, stE1 + 64, y_e1, 0, nullptr, 0);
  k_stats_fin<<<1, 256, 0, stream>>>(stE1, stE1 + 64, stE1 + 128, stE1 + 192, 64,
                                     (float)BNK);
  // 11) BN + LIF on edge conv1 -> s_e1
  k_lif_e1<<<(BNK + 255) / 256, 256, 0, stream>>>(y_e1, stE1 + 128, stE1 + 192,
                                                  g1, be1, s_e1,
                                                  (long long)BNK * 64);
  // 12) edge conv2 (64->128): pass1 stats
  gemm_wmma_wg<TSTEPS, MODE_STATS><<<dim3(BNK / 16, 1), 256, 0, stream>>>(
      e2h, s_e1, (long long)BNK * 64, 64, 128, nullptr, nullptr, nullptr,
      nullptr, nullptr, stE2 + 0, stE2 + 512, nullptr, 0, nullptr, 0);
  k_stats_fin<<<2, 256, 0, stream>>>(stE2, stE2 + 512, stE2 + 1024, stE2 + 1536,
                                     512, (float)BNK);
  // 13) edge conv2 pass2: BN + LIF + max over k -> h2 (T,BN,128)
  gemm_wmma_wg<TSTEPS, MODE_BN_LIF_MAXSTORE>
      <<<dim3(BNK / 16, 1), 256, 0, stream>>>(
      e2h, s_e1, (long long)BNK * 64, 64, 128, nullptr, stE2 + 1024, stE2 + 1536,
      g2, be2, nullptr, nullptr, h2, (long long)BN * 128, nullptr, 0);
  // 14) conv3 (128->256): pass1 stats
  gemm_wmma_wg<TSTEPS, MODE_STATS><<<dim3(BN / 16, 2), 256, 0, stream>>>(
      w3eh, h2, (long long)BN * 128, 128, 256, nullptr, nullptr, nullptr,
      nullptr, nullptr, stW3 + 0, stW3 + 1024, nullptr, 0, nullptr, 0);
  k_stats_fin<<<4, 256, 0, stream>>>(stW3, stW3 + 1024, stW3 + 2048, stW3 + 3072,
                                     1024, (float)BN);
  // 15) conv3 pass2: BN + LIF -> s3
  gemm_wmma_wg<TSTEPS, MODE_BN_LIF_STORE><<<dim3(BN / 16, 2), 256, 0, stream>>>(
      w3eh, h2, (long long)BN * 128, 128, 256, nullptr, stW3 + 2048, stW3 + 3072,
      g3, be3, nullptr, nullptr, s3, (long long)BN * 256, nullptr, 0);
  // 16) conv4 (256->1024): pass1 stats
  gemm_wmma_wg<TSTEPS, MODE_STATS><<<dim3(BN / 16, 8), 256, 0, stream>>>(
      w4h, s3, (long long)BN * 256, 256, 1024, nullptr, nullptr, nullptr,
      nullptr, nullptr, stW4 + 0, stW4 + 4096, nullptr, 0, nullptr, 0);
  k_stats_fin<<<16, 256, 0, stream>>>(stW4, stW4 + 4096, stW4 + 8192,
                                      stW4 + 12288, 4096, (float)BN);
  // 17) conv4 pass2: BN + LIF + max over n -> hfinal
  gemm_wmma_wg<TSTEPS, MODE_BN_LIF_MAXAT><<<dim3(BN / 16, 8), 256, 0, stream>>>(
      w4h, s3, (long long)BN * 256, 256, 1024, nullptr, stW4 + 8192,
      stW4 + 12288, g4, be4, nullptr, nullptr, nullptr, 0, hfinal, NPTS);
  // 18) mean over t -> output (B,1024)
  k_final<<<(BATCH * 1024 + 255) / 256, 256, 0, stream>>>(hfinal, out);
}
